// FlipLinear_15058155339890
// MI455X (gfx1250) — matrix operand order repaired and checked by reading, not compiled, not run on hardware
//
#include <hip/hip_runtime.h>
#include <hip/hip_bf16.h>

// ---------------------------------------------------------------------------
// FlipLinear: out = x @ W^T, W = ternary * per-128-group scale.
// M=8192, N=4096, K=4096. Compute-bound (AI ~1.7 kFLOP/B) -> bf16 WMMA f32 acc.
// GEMM uses the CDNA5 Tensor Data Mover (tensor_load_to_lds) for global->LDS,
// double-buffered, so the VALU/VGPR budget is left to the matrix pipe.
// ---------------------------------------------------------------------------

#define OUT_F 4096
#define IN_F  4096
#define M_TOT 8192

typedef __bf16 bf16_t;
typedef __bf16 v8bf  __attribute__((ext_vector_type(8)));
typedef __bf16 v16bf __attribute__((ext_vector_type(16)));
typedef float  v8f   __attribute__((ext_vector_type(8)));
typedef unsigned int u32x4 __attribute__((ext_vector_type(4)));
typedef int          i32x8 __attribute__((ext_vector_type(8)));
typedef int          i32x4 __attribute__((ext_vector_type(4)));

__device__ __forceinline__ unsigned int f2bf_rne(float f) {
  unsigned int u = __float_as_uint(f);
  return (u + 0x7FFFu + ((u >> 16) & 1u)) >> 16;   // round-to-nearest-even
}
__device__ __forceinline__ unsigned int pack2bf(float a, float b) {
  return f2bf_rne(a) | (f2bf_rne(b) << 16);
}

// --- Kernel 1: x fp32 -> bf16 (8 elements / thread) ------------------------
__global__ __launch_bounds__(256) void cvt_x_bf16(const float* __restrict__ x,
                                                  unsigned int* __restrict__ o) {
  size_t i = ((size_t)blockIdx.x * 256u + threadIdx.x) * 8u;
  float4 f0 = *(const float4*)(x + i);
  float4 f1 = *(const float4*)(x + i + 4);
  uint4 r;
  r.x = pack2bf(f0.x, f0.y);
  r.y = pack2bf(f0.z, f0.w);
  r.z = pack2bf(f1.x, f1.y);
  r.w = pack2bf(f1.z, f1.w);
  *(uint4*)(o + (i >> 1)) = r;
}

// --- Kernel 2: ternary int32 * group scale -> bf16 W -----------------------
__global__ __launch_bounds__(256) void dequant_w_bf16(const int* __restrict__ t,
                                                      const float* __restrict__ s,
                                                      unsigned int* __restrict__ o) {
  size_t i = ((size_t)blockIdx.x * 256u + threadIdx.x) * 8u;
  float sc = s[i >> 7];              // 8 | 128 => one scale per chunk
  int4 t0 = *(const int4*)(t + i);
  int4 t1 = *(const int4*)(t + i + 4);
  uint4 r;
  r.x = pack2bf((float)t0.x * sc, (float)t0.y * sc);
  r.y = pack2bf((float)t0.z * sc, (float)t0.w * sc);
  r.z = pack2bf((float)t1.x * sc, (float)t1.y * sc);
  r.w = pack2bf((float)t1.z * sc, (float)t1.w * sc);
  *(uint4*)(o + (i >> 1)) = r;
}

// --- Kernel 3: bf16 WMMA GEMM with TDM double buffering --------------------
#define BM 128
#define BN 128
#define BK 64
#define LDT (BK + 8)               // 72-half row stride (TDM pad reproduces it):
                                   // r*36 banks mod 64 distinct over 16 rows
#define HALVES_T (BM * LDT)        // 9216 halves per tile buffer
#define STAGE_H  (2 * HALVES_T)    // A + B per stage
#define LDS_BYTES (2 * STAGE_H * 2)  // double buffered: 73728 B

// Issue one TDM 2D tile load: tile 64 halves x 128 rows, row stride 4096 halves,
// bf16 (data_size=2B), LDS padding 4 DWORDs every 32 DWORDs (=> 72-half stride).
// D# packing per CDNA5 ISA (groups 0/1; 2D => rest 0).
__device__ __forceinline__ void tdm_load_tile(unsigned lds_byte_off,
                                              const bf16_t* gptr,
                                              unsigned tensor_rows) {
  unsigned long long ga = (unsigned long long)(uintptr_t)gptr;
  u32x4 g0;
  g0[0] = 1u;                                    // count=1 (valid), user mode
  g0[1] = lds_byte_off;                          // lds_addr (bytes)
  g0[2] = (unsigned)(ga & 0xffffffffu);          // global_addr[31:0]
  g0[3] = (unsigned)((ga >> 32) & 0x01ffffffu)   // global_addr[56:32]
          | (2u << 30);                          // type=2 ("image")
  i32x8 g1;
  g1[0] = (1 << 16)                              // data_size = 2 bytes
        | (1 << 20)                              // pad_enable
        | (4 << 22)                              // pad_interval: 32 DWORDs
        | (3 << 25);                             // pad_amount:   4 DWORDs
  g1[1] = (int)(4096u << 16);                    // tensor_dim0 = 4096 (lo 16)
  g1[2] = (int)(tensor_rows << 16);              // dim0 hi=0 | tensor_dim1 lo
  g1[3] = (int)((tensor_rows >> 16) | (64u << 16)); // dim1 hi | tile_dim0 = 64
  g1[4] = 128;                                   // tile_dim1 = 128, tile_dim2 = 0
  g1[5] = 4096;                                  // tensor_dim0_stride lo
  g1[6] = 0;                                     // stride hi | dim1_stride lo
  g1[7] = 0;
  i32x4 z4 = {0, 0, 0, 0};
  i32x8 z8 = {0, 0, 0, 0, 0, 0, 0, 0};
  __builtin_amdgcn_tensor_load_to_lds(g0, g1, z4, z4, z8, 0);
}

__global__ __launch_bounds__(128) void ternary_gemm_wmma(
    const bf16_t* __restrict__ A,   // [M_TOT][IN_F] bf16 activations
    const bf16_t* __restrict__ B,   // [OUT_F][IN_F] bf16 dequantized weights
    float* __restrict__ C)          // [M_TOT][OUT_F]
{
  extern __shared__ __align__(16) bf16_t smem[];   // 2 stages x (A|B)

  const int tid  = threadIdx.x;
  const int lane = tid & 31;
  const int wave = tid >> 5;

  const int bm = blockIdx.y * BM;
  const int bn = blockIdx.x * BN;

  // 4 waves in 2x2; each wave owns a 64x64 tile = 4x4 WMMA tiles
  const int wm = (wave & 1) << 6;
  const int wn = (wave >> 1) << 6;

  const int lrow = lane & 15;            // A: row-in-tile / B: col-in-tile
  const int lka  = (lane >> 4) << 3;     // A K-chunks at lka and lka+16
  const int lkb  = (lane >> 4) << 3;     // B K-chunks at lkb and lkb+16

  const unsigned ldsBase = (unsigned)(unsigned long long)(uintptr_t)smem;

  v8f acc[4][4] = {};

  const int nk = IN_F / BK;              // 64 stages

  // Preload stage 0 (wave 0 drives the TDM; EXEC-independent DMA)
  if (wave == 0) {
    tdm_load_tile(ldsBase,                 A + (size_t)bm * IN_F, M_TOT);
    tdm_load_tile(ldsBase + HALVES_T * 2,  B + (size_t)bn * IN_F, OUT_F);
  }

  for (int ik = 0; ik < nk; ++ik) {
    const int buf = ik & 1;

    // Kick next stage into the other buffer (its readers finished at the
    // trailing barrier of iteration ik-1).
    if (wave == 0) {
      if (ik + 1 < nk) {
        unsigned sb = (unsigned)(((ik + 1) & 1) * (STAGE_H * 2));
        tdm_load_tile(ldsBase + sb,
                      A + (size_t)bm * IN_F + (size_t)(ik + 1) * BK, M_TOT);
        tdm_load_tile(ldsBase + sb + HALVES_T * 2,
                      B + (size_t)bn * IN_F + (size_t)(ik + 1) * BK, OUT_F);
        __builtin_amdgcn_s_wait_tensorcnt(2);   // current stage's 2 ops done
      } else {
        __builtin_amdgcn_s_wait_tensorcnt(0);   // last stage fully landed
      }
    }
    __syncthreads();                            // data ready for all waves

    const bf16_t* sAp = smem + buf * STAGE_H;
    const bf16_t* sBp = sAp + HALVES_T;

#pragma unroll
    for (int k2 = 0; k2 < BK; k2 += 32) {
      v16bf af[4];
      v16bf bfv[4];
#pragma unroll
      for (int tm = 0; tm < 4; ++tm) {
        // A 16x32 frag: lane holds row lane%16, K=[lka..+7] and [lka+16..+23]
        const bf16_t* p = &sAp[(wm + tm * 16 + lrow) * LDT + k2 + lka];
        v8bf lo = *(const v8bf*)p;
        v8bf hi = *(const v8bf*)(p + 16);
        af[tm] = __builtin_shufflevector(lo, hi,
                 0,1,2,3,4,5,6,7,8,9,10,11,12,13,14,15);
      }
#pragma unroll
      for (int tn = 0; tn < 4; ++tn) {
        // B 32x16 frag: lane holds col lane%16, K=[lkb..+7] and [lkb+16..+23]
        const bf16_t* p = &sBp[(wn + tn * 16 + lrow) * LDT + k2 + lkb];
        v8bf lo = *(const v8bf*)p;
        v8bf hi = *(const v8bf*)(p + 16);
        bfv[tn] = __builtin_shufflevector(lo, hi,
                  0,1,2,3,4,5,6,7,8,9,10,11,12,13,14,15);
      }
#pragma unroll
      for (int tm = 0; tm < 4; ++tm)
#pragma unroll
        for (int tn = 0; tn < 4; ++tn)
          acc[tm][tn] = __builtin_amdgcn_wmma_f32_16x16x32_bf16(
              false, af[tm], false, bfv[tn], (short)0, acc[tm][tn],
              false, false);
    }
    __syncthreads();   // all reads of this buffer done -> TDM may overwrite
  }

  // C layout: lane L, vgpr v -> (row = 8*(L/16)+v, col = L%16) of 16x16 tile
  const int crow0 = bm + wm + ((lane >> 4) << 3);
  const int ccol0 = bn + wn + lrow;
#pragma unroll
  for (int tm = 0; tm < 4; ++tm)
#pragma unroll
    for (int tn = 0; tn < 4; ++tn) {
      float* cp = C + (size_t)(crow0 + tm * 16) * OUT_F + ccol0 + tn * 16;
#pragma unroll
      for (int v = 0; v < 8; ++v)
        cp[(size_t)v * OUT_F] = acc[tm][tn][v];
    }
}

// ---------------------------------------------------------------------------
extern "C" void kernel_launch(void* const* d_in, const int* in_sizes, int n_in,
                              void* d_out, int out_size, void* d_ws, size_t ws_size,
                              hipStream_t stream) {
  (void)in_sizes; (void)n_in; (void)out_size; (void)ws_size;
  const float* x      = (const float*)d_in[0];
  const int*   tern   = (const int*)d_in[1];
  const float* scales = (const float*)d_in[2];

  // workspace: [0, 64MB) = x in bf16 ; [64MB, 96MB) = W in bf16
  unsigned short* xb = (unsigned short*)d_ws;
  unsigned short* wb = (unsigned short*)((char*)d_ws + (size_t)M_TOT * IN_F * 2);

  cvt_x_bf16<<<(M_TOT * IN_F) / (8 * 256), 256, 0, stream>>>(
      x, (unsigned int*)xb);
  dequant_w_bf16<<<(OUT_F * IN_F) / (8 * 256), 256, 0, stream>>>(
      tern, scales, (unsigned int*)wb);

  dim3 grid(OUT_F / BN, M_TOT / BM);   // (32, 64)
  ternary_gemm_wmma<<<grid, 128, LDS_BYTES, stream>>>(
      (const bf16_t*)xb, (const bf16_t*)wb, (float*)d_out);
}
